// Seq2SeqTcr_83476984365335
// MI455X (gfx1250) — matrix operand order repaired and checked by reading, not compiled, not run on hardware
//
#include <hip/hip_runtime.h>

// ---------------------------------------------------------------------------
// Seq2Seq TCR encoder + GRU-attention decoder for MI455X (gfx1250, wave32).
// All large GEMMs (conv-as-im2col, GRU recurrent, FC head) run through
// v_wmma_f32_16x16x32_bf16 with f32 accumulation. Activations/weights are
// bf16 and L2-resident (~40 MB << 192 MB L2): compute-bound, so the goal is
// keeping the WMMA pipe fed — staging is pure b128 vector loads, no
// per-element integer division.
// ---------------------------------------------------------------------------

typedef __attribute__((ext_vector_type(16))) __bf16 v16bf;
typedef __attribute__((ext_vector_type(8)))  float  v8f;
typedef __attribute__((ext_vector_type(4)))  unsigned int u32x4;
typedef unsigned short u16;

union Frag {
    v16bf v;
    u32x4 q[2];
};

__device__ __forceinline__ u16 f2bf(float f) {
    unsigned int u = __float_as_uint(f);
    unsigned int r = u + 0x7FFFu + ((u >> 16) & 1u);   // round-to-nearest-even
    return (u16)(r >> 16);
}
__device__ __forceinline__ float bf2f(u16 s) {
    return __uint_as_float(((unsigned int)s) << 16);
}
__device__ __forceinline__ v8f wmma_bf16(const Frag& a, const Frag& b, v8f c) {
    return __builtin_amdgcn_wmma_f32_16x16x32_bf16(
        /*neg_a=*/false, a.v, /*neg_b=*/false, b.v,
        /*c_mod=*/(short)0, c, /*reuse_a=*/false, /*reuse_b=*/false);
}

// ---------------------------------------------------------------------------
// Unlaunched compile probe for the CDNA5 Tensor Data Mover path:
// zero-filled D# (count=0 -> NULL tensor) + s_wait_tensorcnt.
// ---------------------------------------------------------------------------
#if __has_builtin(__builtin_amdgcn_tensor_load_to_lds)
typedef __attribute__((ext_vector_type(4))) unsigned int tdm_u4;
typedef __attribute__((ext_vector_type(8))) int tdm_i8;
typedef __attribute__((ext_vector_type(4))) int tdm_i4;
__global__ void tdm_probe_kernel() {
    tdm_u4 g0 = {0, 0, 0, 0};
    tdm_i8 g1 = {0, 0, 0, 0, 0, 0, 0, 0};
    tdm_i4 g2 = {0, 0, 0, 0};
    tdm_i4 g3 = {0, 0, 0, 0};
#if defined(__clang_major__) && (__clang_major__ >= 23)
    tdm_i8 g4 = {0, 0, 0, 0, 0, 0, 0, 0};
    __builtin_amdgcn_tensor_load_to_lds(g0, g1, g2, g3, g4, 0);
#else
    __builtin_amdgcn_tensor_load_to_lds(g0, g1, g2, g3, 0);
#endif
    __builtin_amdgcn_s_wait_tensorcnt((short)0);
}
#endif

// ---------------------------------------------------------------------------
// BatchNorm statistics: one block per channel (80 blocks), reduce 32*1600 vals.
// ---------------------------------------------------------------------------
__global__ void bn_stats_kernel(const float* __restrict__ x,
                                const float* __restrict__ gamma,
                                const float* __restrict__ beta,
                                float* __restrict__ scale,
                                float* __restrict__ shift) {
    __shared__ float s1[256], s2[256];
    int c = blockIdx.x, tid = threadIdx.x;
    float a = 0.f, b = 0.f;
    for (int i = tid; i < 32 * 1600; i += 256) {
        float v = x[i * 80 + c];
        a += v; b += v * v;
    }
    s1[tid] = a; s2[tid] = b; __syncthreads();
    for (int s = 128; s > 0; s >>= 1) {
        if (tid < s) { s1[tid] += s1[tid + s]; s2[tid] += s2[tid + s]; }
        __syncthreads();
    }
    if (tid == 0) {
        float mean = s1[0] * (1.f / 51200.f);
        float var  = s2[0] * (1.f / 51200.f) - mean * mean;
        float sc = gamma[c] * rsqrtf(var + 1e-3f);
        scale[c] = sc;
        shift[c] = beta[c] - mean * sc;
    }
}

// Normalize and write bf16 with channels padded 80 -> 96 (zeros), so conv0's
// im2col K (5*96=480) has 32-wide chunks that never cross a tap boundary.
__global__ void bn_apply_kernel(const float* __restrict__ x,
                                const float* __restrict__ scale,
                                const float* __restrict__ shift,
                                u16* __restrict__ xn) {   // [32][1600][96]
    int i = blockIdx.x * 256 + threadIdx.x;
    if (i < 32 * 1600 * 96) {
        int c = i % 96;
        int bt = i / 96;
        xn[i] = (c < 80) ? f2bf(x[bt * 80 + c] * scale[c] + shift[c]) : (u16)0;
    }
}

// Conv weights: in [5][Cin][512] f32 -> out [512][5*CinPad] bf16 (zero pad).
__global__ void convw_transpose_kernel(const float* __restrict__ in,
                                       u16* __restrict__ out,
                                       int Cin, int CinPad) {
    int i = blockIdx.x * 256 + threadIdx.x;
    int total = 512 * 5 * CinPad;
    if (i < total) {
        int n = i / (5 * CinPad);
        int rem = i - n * (5 * CinPad);
        int tap = rem / CinPad;
        int c = rem - tap * CinPad;
        out[i] = (c < Cin) ? f2bf(in[((size_t)tap * Cin + c) * 512 + n]) : (u16)0;
    }
}

// out[NP][K] = transpose(in[K][N]) with zero padding N -> NP
__global__ void transpose_pad_kernel(const float* __restrict__ in, u16* __restrict__ out,
                                     int K, int N, int NP) {
    int i = blockIdx.x * 256 + threadIdx.x;
    if (i < NP * K) {
        int n = i / K, k = i - n * K;
        out[i] = (n < N) ? f2bf(in[k * N + n]) : (u16)0;
    }
}

__global__ void pad_bias_kernel(const float* __restrict__ in, float* __restrict__ out,
                                int n, int np) {
    int i = blockIdx.x * 256 + threadIdx.x;
    if (i < np) out[i] = (i < n) ? in[i] : 0.f;
}

__global__ void zero_us_kernel(u16* __restrict__ p, int n) {
    int i = blockIdx.x * 256 + threadIdx.x;
    if (i < n) p[i] = 0;
}

// ---------------------------------------------------------------------------
// Conv1D as im2col GEMM:  M = 32*800 rows of (b, t_out), N = 512, K = 5*CinPad.
// 128 threads (4 waves), 64x64 tile, K chunks of 32 staged in LDS.
// Invariant: CinPad is a multiple of 32, so a chunk lies inside one tap ->
// tap/cbase are uniform per chunk (SALU) and all staging is b128 loads.
// Epilogue: relu(acc + bias) [+ residual], output bf16 row-major [M][512].
// ---------------------------------------------------------------------------
__global__ __launch_bounds__(128)
void conv_gemm_kernel(const u16* __restrict__ Asrc,   // [32][Tin][CinPad] bf16
                      const u16* __restrict__ BwT,    // [512][Ktot] bf16 (n-major)
                      const float* __restrict__ bias, // [512]
                      const u16* __restrict__ Hres,   // [M][512] residual or null
                      u16* __restrict__ Hout,         // [M][512]
                      int CinPad, int Tin, int stride, int padL, int Ktot) {
    __shared__ __attribute__((aligned(16))) u16 As[64 * 32];
    __shared__ __attribute__((aligned(16))) u16 Bs[64 * 32];   // [n][k]

    const int tid = threadIdx.x;
    const int lane = tid & 31, wave = tid >> 5;
    const int mBase = blockIdx.x * 64;
    const int nBase = blockIdx.y * 64;
    const int b0 = mBase / 800;            // uniform (SALU, once)
    const int t0 = mBase - b0 * 800;

    const v8f zf = {0.f, 0.f, 0.f, 0.f, 0.f, 0.f, 0.f, 0.f};
    v8f acc[4] = {zf, zf, zf, zf};

    const int cl = lane & 15;
    const int hk = (lane < 16) ? 0 : 8;    // A fragment K half offset
    const int ko = (lane < 16) ? 0 : 16;   // unused; B fragment also uses hk

    const int kchunks = Ktot >> 5;         // Ktot is a multiple of 32
    for (int kc = 0; kc < kchunks; ++kc) {
        const int k0 = kc << 5;
        const int tap = k0 / CinPad;       // uniform per chunk
        const int cb  = k0 - tap * CinPad; // uniform, multiple of 32
        // ---- stage A (64 rows x 32 k) + B (64 n x 32 k): 2 b128 each/thread
#pragma unroll
        for (int i = 0; i < 2; ++i) {
            int fq = tid + (i << 7);       // quad id 0..255
            int r  = fq >> 2;              // row / n : 0..63
            int q  = (fq & 3) << 3;        // u16 offset of 16B quad in row
            // A: row -> (b, t_out) without division
            int t  = t0 + r;
            int bb = b0 + (t >= 800 ? 1 : 0);
            t     -= (t >= 800 ? 800 : 0);
            int ti = t * stride - padL + tap;
            u32x4 av = {0u, 0u, 0u, 0u};
            if (ti >= 0 && ti < Tin)
                av = *(const u32x4*)&Asrc[((size_t)bb * Tin + ti) * CinPad + cb + q];
            *(u32x4*)&As[(r << 5) + q] = av;
            // B: contiguous K from pre-transposed weights
            *(u32x4*)&Bs[(r << 5) + q] =
                *(const u32x4*)&BwT[(size_t)(nBase + r) * Ktot + k0 + q];
        }
        __syncthreads();

        // ---- fragments + WMMA ----
        Frag a;
        int r = (wave << 4) + cl;
        a.q[0] = *(const u32x4*)&As[r * 32 + hk];
        a.q[1] = *(const u32x4*)&As[r * 32 + hk + 16];
#pragma unroll
        for (int j = 0; j < 4; ++j) {
            Frag bf;
            int n = (j << 4) + cl;
            bf.q[0] = *(const u32x4*)&Bs[n * 32 + hk];
            bf.q[1] = *(const u32x4*)&Bs[n * 32 + hk + 16];
            acc[j] = wmma_bf16(a, bf, acc[j]);
        }
        __syncthreads();
    }

    // ---- epilogue: C/D layout: VGPR v -> row v (+8 for lanes>=16), col = lane&15
    const int rAdd = (lane < 16) ? 0 : 8;
#pragma unroll
    for (int j = 0; j < 4; ++j) {
        int col = nBase + (j << 4) + cl;
        float bv = bias[col];
#pragma unroll
        for (int v = 0; v < 8; ++v) {
            int m = mBase + (wave << 4) + v + rAdd;
            float val = fmaxf(acc[j][v] + bv, 0.f);
            if (Hres) val += bf2f(Hres[m * 512 + col]);
            Hout[m * 512 + col] = f2bf(val);
        }
    }
}

// ---------------------------------------------------------------------------
// One GRU step (single workgroup, 8 waves):
//  mh = h_prev @ W_rec  (M=32, N=1536, K=512; bf16 WMMA)
//  gates (reset_after): mx via one-hot row gather of gru_kernel,
//  h_new = z*h + (1-z)*tanh(xh + r*hh). Writes h_new (bf16) + cat[:,0:512].
// ---------------------------------------------------------------------------
__global__ __launch_bounds__(256)
void gru_step_kernel(u16* __restrict__ hbuf,          // [32][512] bf16 in/out
                     const u16* __restrict__ wrecT,   // [1536][512] bf16 (transposed)
                     const float* __restrict__ gk,    // [1000][1536] f32
                     const float* __restrict__ gbias, // [2][1536]
                     const int* __restrict__ ytrue,   // [32][201]
                     float* __restrict__ mh,          // [32][1536] scratch
                     u16* __restrict__ cat,           // [32][1024]
                     int u) {
    __shared__ __attribute__((aligned(16))) u16 Hs[32 * 512];
    const int tid = threadIdx.x;
    const int lane = tid & 31, wave = tid >> 5;

    for (int i = tid; i < 32 * 512 / 8; i += 256)
        *(u32x4*)&Hs[i << 3] = *(const u32x4*)&hbuf[i << 3];
    __syncthreads();

    const int cl = lane & 15;
    const int hk = (lane < 16) ? 0 : 8;
    const int ko = (lane < 16) ? 0 : 16;
    const int rAdd = (lane < 16) ? 0 : 8;
    const v8f zf = {0.f, 0.f, 0.f, 0.f, 0.f, 0.f, 0.f, 0.f};

    for (int nt = wave; nt < 96; nt += 8) {
        v8f acc0 = zf, acc1 = zf;
        const int n = (nt << 4) + cl;
        for (int kc = 0; kc < 16; ++kc) {
            const int kb = kc << 5;
            Frag b;
            b.q[0] = *(const u32x4*)&wrecT[n * 512 + kb + hk];
            b.q[1] = *(const u32x4*)&wrecT[n * 512 + kb + hk + 16];
            Frag a0, a1;
            a0.q[0] = *(const u32x4*)&Hs[cl * 512 + kb + hk];
            a0.q[1] = *(const u32x4*)&Hs[cl * 512 + kb + hk + 16];
            a1.q[0] = *(const u32x4*)&Hs[(16 + cl) * 512 + kb + hk];
            a1.q[1] = *(const u32x4*)&Hs[(16 + cl) * 512 + kb + hk + 16];
            acc0 = wmma_bf16(a0, b, acc0);
            acc1 = wmma_bf16(a1, b, acc1);
        }
        int col = (nt << 4) + cl;
#pragma unroll
        for (int v = 0; v < 8; ++v) {
            mh[(v + rAdd) * 1536 + col]      = acc0[v];
            mh[(16 + v + rAdd) * 1536 + col] = acc1[v];
        }
    }
    __threadfence();
    __syncthreads();

    // gates
    for (int i = tid; i < 32 * 512; i += 256) {
        int b = i >> 9, d = i & 511;
        int tok = ytrue[b * 201 + u];
        const float* gr = gk + (size_t)tok * 1536;
        float xz = gr[d]        + gbias[d];
        float xr = gr[512 + d]  + gbias[512 + d];
        float xh = gr[1024 + d] + gbias[1024 + d];
        const float* mr = mh + b * 1536;
        float hz = mr[d]        + gbias[1536 + d];
        float hr = mr[512 + d]  + gbias[1536 + 512 + d];
        float hh = mr[1024 + d] + gbias[1536 + 1024 + d];
        float z  = 1.f / (1.f + __expf(-(xz + hz)));
        float r  = 1.f / (1.f + __expf(-(xr + hr)));
        float cd = tanhf(xh + r * hh);
        float hp = bf2f(Hs[i]);
        float hn = z * hp + (1.f - z) * cd;
        u16 hb = f2bf(hn);
        hbuf[i] = hb;
        cat[b * 1024 + d] = hb;
    }
}

// ---------------------------------------------------------------------------
// Attention (one block per batch row): scores = h . enc_t, softmax over T=800,
// ctx = sum alpha_t * enc_t.  Batched GEMV (M=1) -> VALU path; all enc reads
// are b128 (8 bf16) unpacked with shifts; ctx split over 4 thread groups.
// ---------------------------------------------------------------------------
__global__ __launch_bounds__(256)
void attn_kernel(const u16* __restrict__ enc,  // [32][800][512] bf16
                 const u16* __restrict__ hbuf, // [32][512] bf16
                 u16* __restrict__ cat) {      // [32][1024], writes [512:1024)
    __shared__ float sc[800];
    __shared__ float hsh[512];
    __shared__ float red[256];
    __shared__ float cpart[4][512];
    const int b = blockIdx.x, tid = threadIdx.x;

    for (int i = tid; i < 512; i += 256) hsh[i] = bf2f(hbuf[b * 512 + i]);
    __syncthreads();

    // scores
    for (int t = tid; t < 800; t += 256) {
        const u16* er = enc + ((size_t)b * 800 + t) * 512;
        float s = 0.f;
        for (int dq = 0; dq < 64; ++dq) {
            u32x4 pk = *(const u32x4*)&er[dq << 3];
            const float* hp = &hsh[dq << 3];
#pragma unroll
            for (int c4 = 0; c4 < 4; ++c4) {
                float lo = __uint_as_float(pk[c4] << 16);
                float hi = __uint_as_float(pk[c4] & 0xffff0000u);
                s += lo * hp[2 * c4] + hi * hp[2 * c4 + 1];
            }
        }
        sc[t] = s;
    }
    __syncthreads();

    // softmax over T=800
    float lm = -3.4e38f;
    for (int t = tid; t < 800; t += 256) lm = fmaxf(lm, sc[t]);
    red[tid] = lm; __syncthreads();
    for (int s = 128; s > 0; s >>= 1) {
        if (tid < s) red[tid] = fmaxf(red[tid], red[tid + s]);
        __syncthreads();
    }
    float mx = red[0]; __syncthreads();
    float ls = 0.f;
    for (int t = tid; t < 800; t += 256) { float e = __expf(sc[t] - mx); sc[t] = e; ls += e; }
    red[tid] = ls; __syncthreads();
    for (int s = 128; s > 0; s >>= 1) {
        if (tid < s) red[tid] += red[tid + s];
        __syncthreads();
    }
    float inv = 1.f / red[0];
    __syncthreads();

    // ctx: group tq handles t in [tq*200, tq*200+200), 8 d per thread
    {
        const int tq = tid >> 6;            // 0..3
        const int dq = (tid & 63) << 3;     // d base, 0..504
        float a[8];
#pragma unroll
        for (int j = 0; j < 8; ++j) a[j] = 0.f;
        for (int t = tq * 200; t < tq * 200 + 200; ++t) {
            float al = sc[t];
            u32x4 pk = *(const u32x4*)&enc[((size_t)b * 800 + t) * 512 + dq];
#pragma unroll
            for (int c4 = 0; c4 < 4; ++c4) {
                a[2 * c4]     += al * __uint_as_float(pk[c4] << 16);
                a[2 * c4 + 1] += al * __uint_as_float(pk[c4] & 0xffff0000u);
            }
        }
#pragma unroll
        for (int j = 0; j < 8; ++j) cpart[tq][dq + j] = a[j];
    }
    __syncthreads();
    for (int d = tid; d < 512; d += 256) {
        float s = (cpart[0][d] + cpart[1][d] + cpart[2][d] + cpart[3][d]) * inv;
        cat[b * 1024 + 512 + d] = f2bf(s);
    }
}

// ---------------------------------------------------------------------------
// FC head: logits[32][1008] = cat[32][1024] @ fc_w (N padded 1000->1008), WMMA.
// 16 blocks x 4 waves = 64 waves, one N-tile each (63 used).
// ---------------------------------------------------------------------------
__global__ __launch_bounds__(128)
void fc_kernel(const u16* __restrict__ cat,   // [32][1024] bf16
               const u16* __restrict__ wfcT,  // [1008][1024] bf16 (transposed)
               const float* __restrict__ fcb, // [1008]
               float* __restrict__ logits) {  // [32][1008]
    const int tid = threadIdx.x;
    const int lane = tid & 31, wave = tid >> 5;
    const int nt = blockIdx.x * 4 + wave;
    if (nt >= 63) return;

    const int cl = lane & 15;
    const int hk = (lane < 16) ? 0 : 8;
    const int ko = (lane < 16) ? 0 : 16;
    const v8f zf = {0.f, 0.f, 0.f, 0.f, 0.f, 0.f, 0.f, 0.f};
    v8f acc0 = zf, acc1 = zf;
    const int n = (nt << 4) + cl;

    for (int kc = 0; kc < 32; ++kc) {
        const int kb = kc << 5;
        Frag b;
        b.q[0] = *(const u32x4*)&wfcT[n * 1024 + kb + hk];
        b.q[1] = *(const u32x4*)&wfcT[n * 1024 + kb + hk + 16];
        Frag a0, a1;
        a0.q[0] = *(const u32x4*)&cat[cl * 1024 + kb + hk];
        a0.q[1] = *(const u32x4*)&cat[cl * 1024 + kb + hk + 16];
        a1.q[0] = *(const u32x4*)&cat[(16 + cl) * 1024 + kb + hk];
        a1.q[1] = *(const u32x4*)&cat[(16 + cl) * 1024 + kb + hk + 16];
        acc0 = wmma_bf16(a0, b, acc0);
        acc1 = wmma_bf16(a1, b, acc1);
    }
    const int rAdd = (lane < 16) ? 0 : 8;
    const int col = (nt << 4) + cl;
    const float bv = fcb[col];
#pragma unroll
    for (int v = 0; v < 8; ++v) {
        logits[(v + rAdd) * 1008 + col]      = acc0[v] + bv;
        logits[(16 + v + rAdd) * 1008 + col] = acc1[v] + bv;
    }
}

// ---------------------------------------------------------------------------
// Row softmax over 1000 logits, write into [B][U][V] output at step u.
// ---------------------------------------------------------------------------
__global__ __launch_bounds__(256)
void softmax_out_kernel(const float* __restrict__ logits, // [32][1008]
                        float* __restrict__ out, int u) {
    __shared__ float red[256];
    const int b = blockIdx.x, tid = threadIdx.x;
    const float* row = logits + b * 1008;
    float lm = -3.4e38f;
    for (int v = tid; v < 1000; v += 256) lm = fmaxf(lm, row[v]);
    red[tid] = lm; __syncthreads();
    for (int s = 128; s > 0; s >>= 1) {
        if (tid < s) red[tid] = fmaxf(red[tid], red[tid + s]);
        __syncthreads();
    }
    float m = red[0]; __syncthreads();
    float ls = 0.f;
    for (int v = tid; v < 1000; v += 256) ls += __expf(row[v] - m);
    red[tid] = ls; __syncthreads();
    for (int s = 128; s > 0; s >>= 1) {
        if (tid < s) red[tid] += red[tid + s];
        __syncthreads();
    }
    float inv = 1.f / red[0];
    float* orow = out + ((size_t)b * 200 + u) * 1000;
    for (int v = tid; v < 1000; v += 256) orow[v] = __expf(row[v] - m) * inv;
}

// ---------------------------------------------------------------------------
extern "C" void kernel_launch(void* const* d_in, const int* in_sizes, int n_in,
                              void* d_out, int out_size, void* d_ws, size_t ws_size,
                              hipStream_t stream) {
    (void)in_sizes; (void)n_in; (void)out_size; (void)ws_size;
    const float* x       = (const float*)d_in[0];   // [32][1600][80]
    const int*   ytrue   = (const int*)d_in[1];     // [32][201]
    const float* gamma   = (const float*)d_in[2];   // [80]
    const float* beta    = (const float*)d_in[3];   // [80]
    const float* conv0_w = (const float*)d_in[4];   // [5][80][512]
    const float* conv0_b = (const float*)d_in[5];   // [512]
    const float* tcr_w   = (const float*)d_in[6];   // [4][5][512][512]
    const float* tcr_b   = (const float*)d_in[7];   // [4][512]
    const float* gk      = (const float*)d_in[8];   // [1000][1536]
    const float* grk     = (const float*)d_in[9];   // [512][1536]
    const float* gbias   = (const float*)d_in[10];  // [2][1536]
    const float* fc_w    = (const float*)d_in[11];  // [1024][1000]
    const float* fc_b    = (const float*)d_in[12];  // [1000]
    float* out = (float*)d_out;                     // [32][200][1000]

    char* ws = (char*)d_ws;
    size_t o = 0;
    auto alloc = [&](size_t bytes) { size_t r = o; o += (bytes + 255) & ~(size_t)255; return r; };

    float* bnscale = (float*)(ws + alloc(80 * 4));
    float* bnshift = (float*)(ws + alloc(80 * 4));
    u16* xn    = (u16*)(ws + alloc((size_t)32 * 1600 * 96 * 2));    // padded channels
    u16* hA    = (u16*)(ws + alloc((size_t)25600 * 512 * 2));
    u16* hB    = (u16*)(ws + alloc((size_t)25600 * 512 * 2));
    u16* wc0T  = (u16*)(ws + alloc((size_t)512 * 480 * 2));         // [512][5*96]
    u16* wtcT  = (u16*)(ws + alloc((size_t)4 * 512 * 2560 * 2));    // [512][5*512] x4
    u16* wrecT = (u16*)(ws + alloc((size_t)1536 * 512 * 2));
    u16* wfcT  = (u16*)(ws + alloc((size_t)1008 * 1024 * 2));
    float* fcbp   = (float*)(ws + alloc(1008 * 4));
    u16* hcur  = (u16*)(ws + alloc(32 * 512 * 2));
    u16* cat   = (u16*)(ws + alloc(32 * 1024 * 2));
    float* mh     = (float*)(ws + alloc(32 * 1536 * 4));
    float* logits = (float*)(ws + alloc(32 * 1008 * 4));

    // ---- prep: BN stats, bf16 conversions/transposes, h0 = 0 ----
    bn_stats_kernel<<<80, 256, 0, stream>>>(x, gamma, beta, bnscale, bnshift);
    bn_apply_kernel<<<(32 * 1600 * 96 + 255) / 256, 256, 0, stream>>>(x, bnscale, bnshift, xn);
    convw_transpose_kernel<<<(512 * 5 * 96 + 255) / 256, 256, 0, stream>>>(conv0_w, wc0T, 80, 96);
    for (int i = 0; i < 4; ++i)
        convw_transpose_kernel<<<(512 * 5 * 512 + 255) / 256, 256, 0, stream>>>(
            tcr_w + (size_t)i * 5 * 512 * 512, wtcT + (size_t)i * 512 * 2560, 512, 512);
    transpose_pad_kernel<<<(1536 * 512 + 255) / 256, 256, 0, stream>>>(grk, wrecT, 512, 1536, 1536);
    transpose_pad_kernel<<<(1008 * 1024 + 255) / 256, 256, 0, stream>>>(fc_w, wfcT, 1024, 1000, 1008);
    pad_bias_kernel<<<(1008 + 255) / 256, 256, 0, stream>>>(fc_b, fcbp, 1000, 1008);
    zero_us_kernel<<<(32 * 512 + 255) / 256, 256, 0, stream>>>(hcur, 32 * 512);

    // ---- encoder: conv0 (stride 2, pad 1) + 4 residual TCR blocks (pad 2) ----
    dim3 cgrid(400, 8);  // M/64 x N/64
    conv_gemm_kernel<<<cgrid, 128, 0, stream>>>(xn, wc0T, conv0_b, nullptr, hA,
                                                96, 1600, 2, 1, 480);
    u16* src = hA; u16* dst = hB;
    for (int i = 0; i < 4; ++i) {
        conv_gemm_kernel<<<cgrid, 128, 0, stream>>>(src, wtcT + (size_t)i * 512 * 2560,
                                                    tcr_b + i * 512, src, dst,
                                                    512, 800, 1, 2, 2560);
        u16* t = src; src = dst; dst = t;
    }
    const u16* enc = src;  // after 4 swaps: hA

    // ---- decoder: 200 teacher-forced steps ----
    for (int u = 0; u < 200; ++u) {
        gru_step_kernel<<<1, 256, 0, stream>>>(hcur, wrecT, gk, gbias, ytrue, mh, cat, u);
        attn_kernel<<<32, 256, 0, stream>>>(enc, hcur, cat);
        fc_kernel<<<16, 128, 0, stream>>>(cat, wfcT, fcbp, logits);
        softmax_out_kernel<<<32, 256, 0, stream>>>(logits, out, u);
    }
}
